// MoELayer_26843545600107
// MI455X (gfx1250) — matrix operand order repaired and checked by reading, not compiled, not run on hardware
//
#include <hip/hip_runtime.h>
#include <hip/hip_bf16.h>
#include <math.h>

#define DIM     512
#define HID     1024
#define NEXP    8

typedef __attribute__((ext_vector_type(16))) __bf16        v16bf;
typedef __attribute__((ext_vector_type(8)))  float         v8f;
typedef __attribute__((ext_vector_type(4)))  unsigned int  u32x4;
typedef __attribute__((ext_vector_type(4)))  int           i32x4;
typedef __attribute__((ext_vector_type(8)))  int           i32x8;

static __device__ __forceinline__ unsigned short f32_to_bf16_rne(float f) {
  unsigned int u = __builtin_bit_cast(unsigned int, f);
  unsigned int r = u + 0x7FFFu + ((u >> 16) & 1u);
  return (unsigned short)(r >> 16);
}

// ---------------------------------------------------------------- utilities
__global__ void zero_f32_kernel(float* p, long long n) {
  long long i = (long long)blockIdx.x * blockDim.x + threadIdx.x;
  long long s = (long long)gridDim.x * blockDim.x;
  for (; i < n; i += s) p[i] = 0.0f;
}

__global__ void zero_i32_kernel(int* p, int n) {
  int i = threadIdx.x;
  if (i < n) p[i] = 0;
}

__global__ void cvt_bf16_kernel(const float* __restrict__ in,
                                unsigned short* __restrict__ out, long long n) {
  long long i = (long long)blockIdx.x * blockDim.x + threadIdx.x;
  long long s = (long long)gridDim.x * blockDim.x;
  for (; i < n; i += s) out[i] = f32_to_bf16_rne(in[i]);
}

// Pre-swizzle weights [NEXP][Kdim][Ndim] (f32, row-major) into bf16 WMMA
// B-fragment order: out[((e*KT+kt)*NT+nt)*32 + lane][16 elems],
// element el of lane l = W[kt*32 + (el<8 ? el : el+8) + 8*(l>=16)][nt*16 + (l&15)].
__global__ void swz_w_kernel(const float* __restrict__ in,
                             unsigned short* __restrict__ out,
                             int Kdim, int Ndim) {
  const int NT = Ndim >> 4, KT = Kdim >> 5;
  const long long total = (long long)NEXP * Kdim * Ndim;
  long long i = (long long)blockIdx.x * blockDim.x + threadIdx.x;
  long long s = (long long)gridDim.x * blockDim.x;
  for (long long o = i; o < total; o += s) {
    int el = (int)(o & 15);
    int l  = (int)((o >> 4) & 31);
    long long rest = o >> 9;
    int nt = (int)(rest % NT);
    long long rest2 = rest / NT;
    int kt = (int)(rest2 % KT);
    int e  = (int)(rest2 / KT);
    int k = kt * 32 + ((el < 8) ? el : el + 8) + ((l >> 4) << 3);
    int n = nt * 16 + (l & 15);
    out[o] = f32_to_bf16_rne(in[((size_t)e * Kdim + k) * Ndim + n]);
  }
}

// ---------------------------------------------------------------- gating
__global__ __launch_bounds__(256)
void gate_kernel(const float* __restrict__ x, const float* __restrict__ gw,
                 float* __restrict__ cw, int* __restrict__ bucket,
                 int* __restrict__ cnt, int T) {
  __shared__ float gws[DIM * NEXP];
  const int tid = threadIdx.x;
  for (int i = tid; i < DIM * NEXP; i += 256) gws[i] = gw[i];
  __syncthreads();

  const int wave = tid >> 5, lane = tid & 31;
  const int t = blockIdx.x * 8 + wave;
  if (t >= T) return;

  float acc[NEXP];
#pragma unroll
  for (int e = 0; e < NEXP; ++e) acc[e] = 0.0f;

  const float* xr = x + (size_t)t * DIM;
  for (int d = lane; d < DIM; d += 32) {
    float xv = xr[d];
#pragma unroll
    for (int e = 0; e < NEXP; ++e) acc[e] += xv * gws[d * NEXP + e];
  }
#pragma unroll
  for (int off = 16; off > 0; off >>= 1)
#pragma unroll
    for (int e = 0; e < NEXP; ++e) acc[e] += __shfl_xor(acc[e], off, 32);

  if (lane == 0) {
    int a0 = 0;
    for (int e = 1; e < NEXP; ++e) if (acc[e] > acc[a0]) a0 = e;
    int a1 = (a0 == 0) ? 1 : 0;
    for (int e = 0; e < NEXP; ++e)
      if (e != a0 && acc[e] > acc[a1]) a1 = e;
    float ga = 1.0f / (1.0f + __expf(acc[a1] - acc[a0]));
    float gb2 = 1.0f - ga;
    float row[NEXP];
#pragma unroll
    for (int e = 0; e < NEXP; ++e) row[e] = 0.0f;
    row[a0] = ga; row[a1] = gb2;
#pragma unroll
    for (int e = 0; e < NEXP; ++e) cw[(size_t)t * NEXP + e] = row[e];
    int p0 = atomicAdd(&cnt[a0], 1); bucket[a0 * T + p0] = t;
    int p1 = atomicAdd(&cnt[a1], 1); bucket[a1 * T + p1] = t;
  }
}

__global__ void loss_kernel(const int* __restrict__ cnt, float* loss_out, int T) {
  if (threadIdx.x == 0) {
    float s = 0.0f;
    for (int e = 0; e < NEXP; ++e) {
      float usage = (float)cnt[e] / (float)T;
      float d = usage - (1.0f / NEXP);
      s += d * d;
    }
    *loss_out = (s / NEXP) * NEXP * 0.01f;
  }
}

// ---------------------------------------------------------------- fused FFN
// CTA = 16 routed tokens of expert blockIdx.y; 8 waves.
// X tile gathered by the Tensor Data Mover (gather-mode D#, 16 row indices),
// B fragments stream from pre-swizzled weights (2x global_load_b128 each),
// A fragments from LDS fragment buffers (2x ds_load_b128 each).
__global__ __launch_bounds__(256)
void moe_ffn_kernel(const unsigned short* __restrict__ xb,
                    const unsigned short* __restrict__ w1f,
                    const float* __restrict__ b1,
                    const unsigned short* __restrict__ w2f,
                    const float* __restrict__ b2,
                    const float* __restrict__ cw,
                    const int* __restrict__ bucket,
                    const int* __restrict__ cnt,
                    float* __restrict__ out, int T) {
  __shared__ __align__(32) unsigned short Xrow[16 * DIM];              // 16 KB (TDM dest)
  __shared__ __align__(32) unsigned short XsF[(DIM / 32) * 32 * 16];   // 16 KB
  __shared__ __align__(32) unsigned short HsF[(HID / 32) * 32 * 16];   // 32 KB
  __shared__ int   tok[16];
  __shared__ float cwv[16];

  const int e     = blockIdx.y;
  const int count = cnt[e];
  const int base  = blockIdx.x * 16;
  if (base >= count) return;

  const int tid   = threadIdx.x;
  const int wave  = tid >> 5;
  const int lane  = tid & 31;
  const int laneN = lane & 15;
  const int mOff  = (lane >= 16) ? 8 : 0;

  if (tid < 16) {
    int idx = base + tid;
    int cl  = idx < count ? idx : (count - 1);
    int t   = bucket[e * T + cl];
    tok[tid] = t;
    cwv[tid] = (idx < count) ? cw[(size_t)t * NEXP + e] : 0.0f;
  }
  __syncthreads();

  // ---- TDM gather: 16 token rows (512 x bf16 each) -> Xrow, one DMA op.
  if (wave == 0) {
    int tv = tok[laneN];  // lane i holds token index for row i&15
    unsigned pk[8];
#pragma unroll
    for (int j = 0; j < 8; ++j) {
      unsigned lo = (unsigned)__builtin_amdgcn_readlane(tv, 2 * j) & 0xFFFFu;
      unsigned hi = (unsigned)__builtin_amdgcn_readlane(tv, 2 * j + 1) & 0xFFFFu;
      pk[j] = lo | (hi << 16);
    }
    unsigned long long ga = (unsigned long long)(uintptr_t)xb;
    unsigned ldsoff = (unsigned)(uintptr_t)&Xrow[0];

    // D# group 0: count=1, gather_mode=1, 16-bit indices, lds/global addr, type=2
    u32x4 g0;
    g0[0] = 0x80000001u;                       // gather_mode<<31 | count=1
    g0[1] = ldsoff;                            // lds_addr
    g0[2] = (unsigned)(ga & 0xFFFFFFFFull);    // global_addr[31:0]
    g0[3] = (unsigned)((ga >> 32) & 0x1FFFFFFull) | (2u << 30);  // addr[56:32], type=2

    // D# group 1: data_size=2B, tensor_dim0=DIM, tensor_dim1=T rows,
    //             tile_dim0=DIM, tile_dim1=16 indices, dim0_stride=DIM
    i32x8 g1;
    g1[0] = (int)(1u << 16);                               // data_size=1 (2 bytes)
    g1[1] = (int)(((unsigned)DIM & 0xFFFFu) << 16);        // tensor_dim0 lo16 @ [63:48]
    g1[2] = (int)((((unsigned)DIM >> 16) & 0xFFFFu) |
                  (((unsigned)T & 0xFFFFu) << 16));        // dim0 hi16 | dim1 lo16
    g1[3] = (int)((((unsigned)T >> 16) & 0xFFFFu) |
                  (((unsigned)DIM) << 16));                // dim1 hi16 | tile_dim0
    g1[4] = 16;                                            // tile_dim1 = #indices
    g1[5] = DIM;                                           // tensor_dim0_stride lo32
    g1[6] = 0;
    g1[7] = 0;

    i32x4 g2, g3;
#pragma unroll
    for (int j = 0; j < 4; ++j) { g2[j] = (int)pk[j]; g3[j] = (int)pk[j + 4]; }

#if __clang_major__ >= 23
    i32x8 gz = {0, 0, 0, 0, 0, 0, 0, 0};
    __builtin_amdgcn_tensor_load_to_lds(g0, g1, g2, g3, gz, 0);
#else
    __builtin_amdgcn_tensor_load_to_lds(g0, g1, g2, g3, 0);
#endif
    __builtin_amdgcn_s_wait_tensorcnt(0);
  }
  __syncthreads();

  // LDS->LDS swizzle: row-major Xrow -> A-fragment layout XsF
  for (int i = tid; i < (DIM / 32) * 32 * 16; i += 256) {
    int el = i & 15, l = (i >> 4) & 31, kt = i >> 9;
    int k  = kt * 32 + ((el < 8) ? el : el + 8) + ((l >= 16) ? 8 : 0);
    XsF[i] = Xrow[(l & 15) * DIM + k];
  }
  __syncthreads();

  const v8f vzero = {0.f, 0.f, 0.f, 0.f, 0.f, 0.f, 0.f, 0.f};

  // ---- stage 1: H = gelu(X @ W1 + b1); wave owns hidden cols [128w,128w+128)
  {
    const unsigned short* w1p = w1f + (size_t)e * (DIM / 32) * (HID / 16) * 512;
    v8f acc[8];
#pragma unroll
    for (int nt = 0; nt < 8; ++nt) acc[nt] = vzero;

    for (int kt = 0; kt < DIM / 32; ++kt) {
      v16bf a = *reinterpret_cast<const v16bf*>(&XsF[(kt * 32 + lane) * 16]);
      const unsigned short* wk =
          w1p + ((size_t)kt * (HID / 16) + wave * 8) * 512 + lane * 16;
      if (kt + 1 < DIM / 32)
        __builtin_prefetch(wk + (size_t)(HID / 16) * 512, 0, 0);
#pragma unroll
      for (int nt = 0; nt < 8; ++nt) {
        v16bf b = *reinterpret_cast<const v16bf*>(wk + nt * 512);
        acc[nt] = __builtin_amdgcn_wmma_f32_16x16x32_bf16(
            false, a, false, b, (short)0, acc[nt], false, false);
      }
    }
    // bias + exact gelu; scatter into stage-2 A-fragment layout
#pragma unroll
    for (int nt = 0; nt < 8; ++nt) {
      const int col = wave * 128 + nt * 16 + laneN;
      const float bb = b1[e * HID + col];
      const int kt2 = col >> 5;
      const int kk  = col & 31;
      const int hi2 = (kk >> 3) & 1;
      const int el2 = (kk < 16) ? (kk & 7) : ((kk & 7) + 8);
#pragma unroll
      for (int r = 0; r < 8; ++r) {
        const int M = r + mOff;
        float h = acc[nt][r] + bb;
        float g = 0.5f * h * (1.0f + erff(h * 0.70710678118654752f));
        HsF[(kt2 * 32 + (M + hi2 * 16)) * 16 + el2] = f32_to_bf16_rne(g);
      }
    }
  }
  __syncthreads();

  // ---- stage 2: out += (H @ W2 + b2) * cw; wave owns out cols [64w,64w+64)
  {
    const unsigned short* w2p = w2f + (size_t)e * (HID / 32) * (DIM / 16) * 512;
    v8f acc[4];
#pragma unroll
    for (int nt = 0; nt < 4; ++nt) acc[nt] = vzero;

    for (int kt = 0; kt < HID / 32; ++kt) {
      v16bf a = *reinterpret_cast<const v16bf*>(&HsF[(kt * 32 + lane) * 16]);
      const unsigned short* wk =
          w2p + ((size_t)kt * (DIM / 16) + wave * 4) * 512 + lane * 16;
      if (kt + 1 < HID / 32)
        __builtin_prefetch(wk + (size_t)(DIM / 16) * 512, 0, 0);
#pragma unroll
      for (int nt = 0; nt < 4; ++nt) {
        v16bf b = *reinterpret_cast<const v16bf*>(wk + nt * 512);
        acc[nt] = __builtin_amdgcn_wmma_f32_16x16x32_bf16(
            false, a, false, b, (short)0, acc[nt], false, false);
      }
    }
#pragma unroll
    for (int nt = 0; nt < 4; ++nt) {
      const int col = wave * 64 + nt * 16 + laneN;
      const float bb = b2[e * DIM + col];
#pragma unroll
      for (int r = 0; r < 8; ++r) {
        const int M = r + mOff;
        float v = (acc[nt][r] + bb) * cwv[M];
        atomicAdd(&out[(size_t)tok[M] * DIM + col], v);
      }
    }
  }
}

// ---------------------------------------------------------------- launch
extern "C" void kernel_launch(void* const* d_in, const int* in_sizes, int n_in,
                              void* d_out, int out_size, void* d_ws, size_t ws_size,
                              hipStream_t stream) {
  const float* x      = (const float*)d_in[0];
  const float* gate_w = (const float*)d_in[1];
  const float* w1     = (const float*)d_in[2];
  const float* b1     = (const float*)d_in[3];
  const float* w2     = (const float*)d_in[4];
  const float* b2     = (const float*)d_in[5];
  float* out = (float*)d_out;

  const int T = in_sizes[0] / DIM;   // 32768 tokens

  char* p = (char*)d_ws;
  unsigned short* w1f = (unsigned short*)p; p += (size_t)NEXP * DIM * HID * 2;
  unsigned short* w2f = (unsigned short*)p; p += (size_t)NEXP * HID * DIM * 2;
  unsigned short* xb  = (unsigned short*)p; p += (size_t)T * DIM * 2;
  float* cw   = (float*)p;          p += (size_t)T * NEXP * 4;
  int*   bkt  = (int*)p;            p += (size_t)NEXP * T * 4;
  int*   cnt  = (int*)p;            p += 64;

  zero_f32_kernel<<<1024, 256, 0, stream>>>(out, (long long)T * DIM + 1);
  zero_i32_kernel<<<1, 32, 0, stream>>>(cnt, NEXP);

  swz_w_kernel<<<1024, 256, 0, stream>>>(w1, w1f, DIM, HID);
  swz_w_kernel<<<1024, 256, 0, stream>>>(w2, w2f, HID, DIM);
  cvt_bf16_kernel<<<2048, 256, 0, stream>>>(x, xb, (long long)T * DIM);

  gate_kernel<<<T / 8, 256, 0, stream>>>(x, gate_w, cw, bkt, cnt, T);
  loss_kernel<<<1, 32, 0, stream>>>(cnt, out + (size_t)T * DIM, T);

  dim3 grid((T + 15) / 16, NEXP);
  moe_ffn_kernel<<<grid, 256, 0, stream>>>(xb, w1f, b1, w2f, b2, cw, bkt, cnt, out, T);
}
